// GATConv_85976655331561
// MI455X (gfx1250) — matrix operand order repaired and checked by reading, not compiled, not run on hardware
//
#include <hip/hip_runtime.h>
#include <hip/hip_bf16.h>

typedef __attribute__((ext_vector_type(16))) __bf16 v16bf;
typedef __attribute__((ext_vector_type(8)))  __bf16 v8bf;
typedef __attribute__((ext_vector_type(8)))  float  v8f;
typedef unsigned int u32x4 __attribute__((ext_vector_type(4)));
typedef int i32x8 __attribute__((ext_vector_type(8)));
typedef int i32x4 __attribute__((ext_vector_type(4)));
typedef __attribute__((address_space(3))) unsigned short lds_us;

union ABu { v16bf v; v8bf h[2]; };

#define IN_DIM   256
#define HD       128          // H*D
#define WROWS    272          // 128 lin_l + 128 lin_r + 4 a1 + 4 a2 + 8 pad
#define LDSROW   264          // 256 + 8 pad shorts (TDM pad fields reproduce this)
#define MT       4            // 4 node subtiles of 16 -> 64 nodes per workgroup
#define EPW      128          // edges per wave in the aggregation pass

#if defined(__has_builtin)
#  if __has_builtin(__builtin_amdgcn_tensor_load_to_lds) && \
      __has_builtin(__builtin_amdgcn_s_wait_tensorcnt)
#    define ATH_TDM 1
#  endif
#endif
#ifndef ATH_TDM
#  define ATH_TDM 0
#endif

__device__ __forceinline__ unsigned short f2bf(float f) {
    unsigned u = __float_as_uint(f);
    u += 0x7FFFu + ((u >> 16) & 1u);          // round-to-nearest-even
    return (unsigned short)(u >> 16);
}

// monotonic uint encoding of float for atomicMax-based segment max
__device__ __forceinline__ unsigned enc_f(float v) {
    int iv = __float_as_int(v);
    return (iv >= 0) ? ((unsigned)iv | 0x80000000u) : (unsigned)(~iv);
}
__device__ __forceinline__ float dec_f(unsigned e) {
    int iv = (e & 0x80000000u) ? (int)(e & 0x7FFFFFFFu) : (int)(~e);
    return __int_as_float(iv);
}

// ---------- pass 0: x f32 -> bf16 (streaming, feeds the TDM tile loads) ----------
__global__ void xcvt_kernel(const float* __restrict__ x, unsigned short* __restrict__ xbf,
                            int total8) {
    int i = blockIdx.x * 256 + threadIdx.x;      // unit = 8 floats
    if (i >= total8) return;
    const float4* s = (const float4*)x + (size_t)i * 2;
    float4 v0 = s[0], v1 = s[1];
    ushort4 o0, o1;
    o0.x = f2bf(v0.x); o0.y = f2bf(v0.y); o0.z = f2bf(v0.z); o0.w = f2bf(v0.w);
    o1.x = f2bf(v1.x); o1.y = f2bf(v1.y); o1.z = f2bf(v1.z); o1.w = f2bf(v1.w);
    ushort4* d = (ushort4*)xbf + (size_t)i * 2;
    d[0] = o0; d[1] = o1;
}

// ---------- pass 1: pack all weights into one bf16 matrix [WROWS x 256] ----------
__global__ void wcvt_kernel(const float* __restrict__ a1w, const float* __restrict__ a2w,
                            const float* __restrict__ llw, const float* __restrict__ lrw,
                            unsigned short* __restrict__ Wbf) {
    int idx = blockIdx.x * 256 + threadIdx.x;          // < WROWS*256
    int r = idx >> 8, k = idx & 255;
    float v = 0.f;
    if      (r < 128) v = llw[r * 256 + k];
    else if (r < 256) v = lrw[(r - 128) * 256 + k];
    else if (r < 260) v = a1w[(r - 256) * 256 + k];
    else if (r < 264) v = a2w[(r - 260) * 256 + k];
    Wbf[idx] = f2bf(v);
}

// ---------- pass 2: init segment tables (ws is poisoned each run) ----------
__global__ void seginit_kernel(unsigned* __restrict__ seg_enc, float* __restrict__ seg_sum, int M) {
    int i = blockIdx.x * 256 + threadIdx.x;
    if (i < M) { seg_enc[i] = 0u; seg_sum[i] = 0.f; }
}

// ---------- pass 3: fused node GEMM via WMMA bf16, x-tile staged by TDM ----------
// out cols: tiles 0..7 -> h_l (lin_l), 8..15 -> d_out (lin_r), 16 -> attn_node[0:8]
__global__ __launch_bounds__(256)
void node_gemm_kernel(const unsigned short* __restrict__ xbf,
                      const unsigned short* __restrict__ Wbf,
                      const float* __restrict__ a1b, const float* __restrict__ a2b,
                      const float* __restrict__ llb, const float* __restrict__ lrb,
                      float* __restrict__ attn_node, float* __restrict__ h_l,
                      float* __restrict__ out, int N) {
    __shared__ unsigned short lx[64 * LDSROW];
    const int t  = threadIdx.x;
    const int nb = blockIdx.x * 64;

#if ATH_TDM
    if (t < 32) {   // wave 0 issues one tensor DMA: 64x256 bf16 tile, hw-padded rows
        unsigned lds_base = (unsigned)(unsigned long long)(lds_us*)&lx[0];
        unsigned long long ga =
            (unsigned long long)(const void*)(xbf + (size_t)nb * IN_DIM);
        u32x4 g0;
        g0.x = 1u;                                   // count=1, user descriptor
        g0.y = lds_base;                             // lds_addr (bytes)
        g0.z = (unsigned)ga;                         // global_addr lo
        g0.w = (unsigned)(ga >> 32) | (2u << 30);    // global_addr hi | type=2 (image)
        unsigned dim1 = (unsigned)(N - nb);          // remaining rows: OOB rows read 0
        i32x8 g1;
        g1[0] = (int)((1u << 16)        // data_size = 2 bytes
                    | (1u << 20)        // pad_enable
                    | (6u << 22)        // pad_interval: every 128 DWORDs (= one row)
                    | (3u << 25));      // pad_amount: 4 DWORDs (= 8 shorts) -> stride 264
        g1[1] = (int)(256u << 16);                          // tensor_dim0 = 256 (lo16)
        g1[2] = (int)((dim1 & 0xFFFFu) << 16);              // dim0 hi16=0 | dim1 lo16
        g1[3] = (int)((dim1 >> 16) | (256u << 16));         // dim1 hi16 | tile_dim0 = 256
        g1[4] = 64;                                         // tile_dim1 = 64, tile_dim2 = 0
        g1[5] = 256;                                        // tensor_dim0_stride = 256
        g1[6] = 0; g1[7] = 0;
        i32x4 z4 = {0, 0, 0, 0};
        i32x8 z8 = {0, 0, 0, 0, 0, 0, 0, 0};
        // 6-arg toolchain variant: (g0, g1, g2, g3, extra, cpol)
        __builtin_amdgcn_tensor_load_to_lds(g0, g1, z4, z4, z8, 0);
        __builtin_amdgcn_s_wait_tensorcnt(0);
    }
#else
    for (int id = t; id < 64 * 32; id += 256) {   // fallback: 16B-chunk copy
        int rr = id >> 5, cc = id & 31;
        int node = nb + rr;
        uint4 v;
        if (node < N) v = ((const uint4*)(xbf + (size_t)node * IN_DIM))[cc];
        else { v.x = 0u; v.y = 0u; v.z = 0u; v.w = 0u; }
        *(uint4*)&lx[rr * LDSROW + cc * 8] = v;
    }
#endif
    __syncthreads();

    const int lane = t & 31, wv = t >> 5;
    const int lh = lane >> 4, ln = lane & 15;

    for (int job = wv; job < MT * 17; job += 8) {
        const int mt = job / 17, ct = job % 17;
        float bv;
        if      (ct < 8)  bv = llb[ct * 16 + ln];
        else if (ct < 16) bv = lrb[(ct - 8) * 16 + ln];
        else              bv = (ln < 4) ? a1b[ln] : (ln < 8 ? a2b[ln - 4] : 0.f);
        v8f c;
        #pragma unroll
        for (int i = 0; i < 8; ++i) c[i] = bv;

        const unsigned short* wrow = Wbf + (size_t)(ct * 16 + ln) * IN_DIM;
        const unsigned short* arow = &lx[(mt * 16 + ln) * LDSROW];
        #pragma unroll
        for (int kb = 0; kb < IN_DIM; kb += 32) {
            ABu a, b;
            // A 16x32 bf16 layout: lanes0-15 hold K {kb..kb+7, kb+16..23}; lanes16-31 K {+8,+24}
            a.h[0] = *(const v8bf*)(arow + kb + lh * 8);
            a.h[1] = *(const v8bf*)(arow + kb + lh * 8 + 16);
            // B 32x16 bf16 layout: lanes0-15 hold K {kb..kb+7, kb+16..23}; lanes16-31 K {+8,+24}
            b.h[0] = *(const v8bf*)(wrow + kb + lh * 8);
            b.h[1] = *(const v8bf*)(wrow + kb + lh * 8 + 16);
            c = __builtin_amdgcn_wmma_f32_16x16x32_bf16(false, a.v, false, b.v,
                                                        (short)0, c, false, false);
        }
        #pragma unroll
        for (int r2 = 0; r2 < 8; ++r2) {   // C/D: vgpr r -> M=r (lanes0-15) / M=r+8 (lanes16-31)
            int node = nb + mt * 16 + lh * 8 + r2;
            if (node >= N) continue;
            if      (ct < 8)  h_l[(size_t)node * HD + ct * 16 + ln] = c[r2];
            else if (ct < 16) out[(size_t)node * HD + (ct - 8) * 16 + ln] = c[r2];
            else if (ln < 8)  attn_node[(size_t)node * 8 + ln] = c[r2];
        }
    }
}

// ---------- pass 4: edge logits + segment max ----------
__global__ void edge_logit_kernel(const int* __restrict__ row, const int* __restrict__ col,
                                  const float* __restrict__ attn_node,
                                  float* __restrict__ attn_edge, unsigned* __restrict__ seg_enc,
                                  int E4) {
    int idx = blockIdx.x * 256 + threadIdx.x;
    if (idx >= E4) return;
    int e = idx >> 2, h = idx & 3;
    int r = row[e], c = col[e];
    float v = attn_node[(size_t)r * 8 + h] + attn_node[(size_t)c * 8 + 4 + h];
    attn_edge[idx] = v;
    atomicMax(&seg_enc[(size_t)r * 4 + h], enc_f(v));
}

// ---------- pass 5: decode max; isolated nodes -> 0 ----------
__global__ void segfix_kernel(const unsigned* __restrict__ seg_enc, float* __restrict__ seg_val, int M) {
    int i = blockIdx.x * 256 + threadIdx.x;
    if (i >= M) return;
    float m = dec_f(seg_enc[i]);
    if (!(m >= -3.0e38f && m <= 3.0e38f)) m = 0.f;   // NaN/inf/untouched -> 0
    seg_val[i] = m;
}

// ---------- pass 6: exp + segment sum ----------
__global__ void edge_exp_kernel(const int* __restrict__ row, float* __restrict__ attn_edge,
                                const float* __restrict__ seg_val, float* __restrict__ seg_sum,
                                int E4) {
    int idx = blockIdx.x * 256 + threadIdx.x;
    if (idx >= E4) return;
    int e = idx >> 2, h = idx & 3;
    int r = row[e];
    float v = __expf(attn_edge[idx] - seg_val[(size_t)r * 4 + h]);
    attn_edge[idx] = v;
    atomicAdd(&seg_sum[(size_t)r * 4 + h], v);
}

// ---------- pass 7: reciprocal of sums ----------
__global__ void segrcp_kernel(float* __restrict__ seg_sum, int M) {
    int i = blockIdx.x * 256 + threadIdx.x;
    if (i < M) seg_sum[i] = 1.f / (seg_sum[i] + 1e-16f);
}

// ---------- pass 8: weighted gather + register-accumulated scatter ----------
// row is sorted: accumulate in regs while dest row unchanged, flush atomics on change.
__global__ __launch_bounds__(256)
void edge_aggr_kernel(const int* __restrict__ row, const int* __restrict__ col,
                      const float* __restrict__ ew, const float* __restrict__ seg_rcp,
                      const float* __restrict__ h_l, float* __restrict__ out, int E) {
    const int lane = threadIdx.x & 31;
    const int wave = blockIdx.x * (blockDim.x >> 5) + (threadIdx.x >> 5);
    int e0 = wave * EPW;
    if (e0 >= E) return;
    int e1 = e0 + EPW; if (e1 > E) e1 = E;
    const int h  = lane >> 3;       // head for this lane's channel block
    const int ch = lane * 4;        // 4 channels per lane, 128 total
    float a0 = 0.f, a1 = 0.f, a2 = 0.f, a3 = 0.f;
    int cur = -1;
    for (int e = e0; e < e1; ++e) {
        int r = row[e], c = col[e];
        if (r != cur) {
            if (cur >= 0) {
                float* o = out + (size_t)cur * HD + ch;
                atomicAdd(o + 0, a0); atomicAdd(o + 1, a1);
                atomicAdd(o + 2, a2); atomicAdd(o + 3, a3);
            }
            a0 = a1 = a2 = a3 = 0.f; cur = r;
        }
        if (e + 8 < e1) {  // pull the random gather line ~8 edges ahead
            int cp = col[e + 8];
            __builtin_prefetch(&h_l[(size_t)cp * HD + ch], 0, 0);
        }
        float w = ew[(size_t)e * 4 + h] * seg_rcp[(size_t)r * 4 + h];
        const float4 xv = *(const float4*)(h_l + (size_t)c * HD + ch);
        a0 = fmaf(w, xv.x, a0); a1 = fmaf(w, xv.y, a1);
        a2 = fmaf(w, xv.z, a2); a3 = fmaf(w, xv.w, a3);
    }
    if (cur >= 0) {
        float* o = out + (size_t)cur * HD + ch;
        atomicAdd(o + 0, a0); atomicAdd(o + 1, a1);
        atomicAdd(o + 2, a2); atomicAdd(o + 3, a3);
    }
}

extern "C" void kernel_launch(void* const* d_in, const int* in_sizes, int n_in,
                              void* d_out, int out_size, void* d_ws, size_t ws_size,
                              hipStream_t stream) {
    const float* x   = (const float*)d_in[0];
    const int*   row = (const int*)  d_in[1];
    const int*   col = (const int*)  d_in[2];
    const float* a1w = (const float*)d_in[3];
    const float* a1b = (const float*)d_in[4];
    const float* a2w = (const float*)d_in[5];
    const float* a2b = (const float*)d_in[6];
    const float* llw = (const float*)d_in[7];
    const float* llb = (const float*)d_in[8];
    const float* lrw = (const float*)d_in[9];
    const float* lrb = (const float*)d_in[10];
    float* out = (float*)d_out;
    const int N = in_sizes[0] / IN_DIM;
    const int E = in_sizes[1];

    char* p = (char*)d_ws;
    auto take = [&](size_t bytes) -> char* {
        char* q = p; p += (bytes + 255) & ~(size_t)255; return q;
    };
    unsigned short* xbf      = (unsigned short*)take((size_t)N * IN_DIM * 2);
    unsigned short* Wbf      = (unsigned short*)take((size_t)WROWS * IN_DIM * 2);
    float*          attnN    = (float*)take((size_t)N * 8 * 4);
    float*          h_l      = (float*)take((size_t)N * HD * 4);
    float*          attnE    = (float*)take((size_t)E * 4 * 4);
    unsigned*       seg_enc  = (unsigned*)take((size_t)N * 4 * 4);
    float*          seg_val  = (float*)take((size_t)N * 4 * 4);
    float*          seg_sum  = (float*)take((size_t)N * 4 * 4);
    (void)ws_size; (void)n_in; (void)out_size;

    const int M      = N * 4;
    const int E4     = E * 4;
    const int total8 = N * (IN_DIM / 8);

    xcvt_kernel<<<(total8 + 255) / 256, 256, 0, stream>>>(x, xbf, total8);
    wcvt_kernel<<<WROWS, 256, 0, stream>>>(a1w, a2w, llw, lrw, Wbf);
    seginit_kernel<<<(M + 255) / 256, 256, 0, stream>>>(seg_enc, seg_sum, M);
    node_gemm_kernel<<<(N + 63) / 64, 256, 0, stream>>>(xbf, Wbf, a1b, a2b, llb, lrb,
                                                        attnN, h_l, out, N);
    edge_logit_kernel<<<(E4 + 255) / 256, 256, 0, stream>>>(row, col, attnN, attnE, seg_enc, E4);
    segfix_kernel<<<(M + 255) / 256, 256, 0, stream>>>(seg_enc, seg_val, M);
    edge_exp_kernel<<<(E4 + 255) / 256, 256, 0, stream>>>(row, attnE, seg_val, seg_sum, E4);
    segrcp_kernel<<<(M + 255) / 256, 256, 0, stream>>>(seg_sum, M);
    const int waves  = (E + EPW - 1) / EPW;
    const int blocks = (waves + 7) / 8;
    edge_aggr_kernel<<<blocks, 256, 0, stream>>>(row, col, attnE, seg_sum, h_l, out, E);
}
